// CaptioningRNN_6751688589710
// MI455X (gfx1250) — matrix operand order repaired and checked by reading, not compiled, not run on hardware
//
#include <hip/hip_runtime.h>
#include <math.h>

// Problem constants (from reference): N=256, T=128, D=512, H=1024, L=16
#define NB   256
#define TT   128
#define DD   512
#define HH   1024
#define KTOT 2560           // D + H + H
#define NKB  80             // KTOT / 32
#define NCOL 4096           // 4*H
#define FRAG_ELEMS 512      // 32x16 bf16 fragment

typedef __attribute__((ext_vector_type(16))) __bf16 v16bf;
typedef __attribute__((ext_vector_type(8)))  float  v8f;
typedef __attribute__((ext_vector_type(4)))  unsigned int u32x4;

union BFrag {
    v16bf bf;
    u32x4 q[2];
};

// round-to-nearest-even fp32 -> bf16 (bit pattern as ushort)
__device__ __forceinline__ unsigned short f2bf(float f) {
    unsigned int u = __float_as_uint(f);
    u += 0x7FFFu + ((u >> 16) & 1u);
    return (unsigned short)(u >> 16);
}

__device__ __forceinline__ float sigm(float x) {
    return 1.0f / (1.0f + __expf(-x));
}

// CDNA5 async global->LDS copy (ASYNCcnt tracked), per-lane 16B.
// dstLds = byte offset of LDS destination (low 32 bits of generic LDS ptr),
// gaddr  = 64-bit global source address.
__device__ __forceinline__ void async_copy_b128(unsigned dstLds,
                                                unsigned long long gaddr) {
    asm volatile("global_load_async_to_lds_b128 %0, %1, off"
                 :
                 : "v"(dstLds), "v"(gaddr)
                 : "memory");
}
__device__ __forceinline__ void wait_asynccnt0() {
    asm volatile("s_wait_asynccnt 0x0" ::: "memory");
}

// ---------------------------------------------------------------------------
// One-time: swizzle [Wx; Wh; Wattn] (2560 x 4096 fp32) into bf16 WMMA-B
// fragment order. Fragment f = kb*256 + cb covers K rows [kb*32,kb*32+32),
// cols [cb*16, cb*16+16). Per-lane layout (contiguous k order used here, 16-bit B):
//   lanes 0-15:  col = lane,    K = 0..15  (2 per VGPR)
//   lanes 16-31: col = lane-16, K = 16..31
// ---------------------------------------------------------------------------
__global__ __launch_bounds__(256) void swz_w_kernel(
    const float* __restrict__ Wx, const float* __restrict__ Wh,
    const float* __restrict__ Wattn, unsigned short* __restrict__ Wsw)
{
    int e = blockIdx.x * 256 + threadIdx.x;           // element in swizzled buf
    if (e >= KTOT * NCOL) return;
    int f    = e >> 9;                                // fragment index
    int r    = e & 511;
    int lane = r >> 4;
    int idx  = r & 15;                                // K offset within lane
    int kb   = f >> 8;
    int cb   = f & 255;
    int klocal = (lane & 16) + idx;                   // +16 for lanes 16..31
    int srcK   = (kb << 5) + klocal;
    int col    = (cb << 4) + (lane & 15);
    float v;
    if (srcK < DD)            v = Wx[(size_t)srcK * NCOL + col];
    else if (srcK < DD + HH)  v = Wh[(size_t)(srcK - DD) * NCOL + col];
    else                      v = Wattn[(size_t)(srcK - DD - HH) * NCOL + col];
    Wsw[e] = f2bf(v);
}

// One-time: x (N,T,D) fp32 -> bf16, same layout
__global__ __launch_bounds__(256) void conv_x_kernel(
    const float* __restrict__ x, unsigned short* __restrict__ xb, int total)
{
    int i = blockIdx.x * 256 + threadIdx.x;
    if (i < total) xb[i] = f2bf(x[i]);
}

// One-time: h0 = c0 = mean over the 16 spatial locations of A
__global__ __launch_bounds__(256) void init_h0_kernel(
    const float* __restrict__ A, float* __restrict__ Hprev,
    float* __restrict__ Cst, unsigned short* __restrict__ Hbf)
{
    int i = blockIdx.x * 256 + threadIdx.x;           // over N*H
    const float* ar = A + (size_t)i * 16;
    float s = 0.f;
#pragma unroll
    for (int l = 0; l < 16; ++l) s += ar[l];
    s *= (1.0f / 16.0f);
    Hprev[i] = s;
    Cst[i]   = s;
    Hbf[i]   = f2bf(s);
}

// ---------------------------------------------------------------------------
// Per-step attention: one block per batch element n.
// scores[l] = (1/32) * sum_h A[n,h,l]*Hprev[n,h]; softmax over 16;
// attn[n,h] = sum_l A[n,h,l]*w[l]  -> bf16.
// Deterministic: shuffle reduce within wave32, fixed-order LDS tree.
// ---------------------------------------------------------------------------
__global__ __launch_bounds__(256) void attn_kernel(
    const float* __restrict__ A, const float* __restrict__ Hprev,
    unsigned short* __restrict__ Attnbf)
{
    __shared__ float wred[8 * 16];
    __shared__ float ssco[16];
    const int n    = blockIdx.x;
    const int tid  = threadIdx.x;
    const int lane = tid & 31;
    const int wave = tid >> 5;

    float s[16];
#pragma unroll
    for (int l = 0; l < 16; ++l) s[l] = 0.f;

#pragma unroll
    for (int j = 0; j < 4; ++j) {
        int hh = j * 256 + tid;
        float hv = Hprev[(size_t)n * HH + hh];
        const float* ar = A + ((size_t)n * HH + hh) * 16;
#pragma unroll
        for (int l = 0; l < 16; ++l) s[l] += ar[l] * hv;
    }
    // wave32 reduction
#pragma unroll
    for (int off = 16; off > 0; off >>= 1) {
#pragma unroll
        for (int l = 0; l < 16; ++l) s[l] += __shfl_down(s[l], off, 32);
    }
    if (lane == 0) {
#pragma unroll
        for (int l = 0; l < 16; ++l) wred[wave * 16 + l] = s[l];
    }
    __syncthreads();
    if (tid < 16) {
        float sc = 0.f;
#pragma unroll
        for (int w = 0; w < 8; ++w) sc += wred[w * 16 + tid];
        ssco[tid] = sc * 0.03125f;                    // 1/sqrt(1024)
    }
    __syncthreads();

    // redundant per-thread softmax over 16 (deterministic, no extra barrier)
    float mx = -3.402823466e38f;
#pragma unroll
    for (int l = 0; l < 16; ++l) mx = fmaxf(mx, ssco[l]);
    float wl[16];
    float sum = 0.f;
#pragma unroll
    for (int l = 0; l < 16; ++l) { wl[l] = __expf(ssco[l] - mx); sum += wl[l]; }
    float inv = 1.0f / sum;
#pragma unroll
    for (int l = 0; l < 16; ++l) wl[l] *= inv;

#pragma unroll
    for (int j = 0; j < 4; ++j) {
        int hh = j * 256 + tid;
        const float* ar = A + ((size_t)n * HH + hh) * 16;
        float av = 0.f;
#pragma unroll
        for (int l = 0; l < 16; ++l) av += ar[l] * wl[l];
        Attnbf[(size_t)n * HH + hh] = f2bf(av);
    }
}

// ---------------------------------------------------------------------------
// Per-step fused GEMM + LSTM:
//   gates = [x_t | h | attn](256x2560) @ Wsw(2560x4096) + b
// Block: 8 waves. Wave w owns batch rows m0 = blockIdx.x*128 + w*16.
// Block owns 64 H-columns (blockIdx.y*64) replicated over the 4 gates, so the
// i/f/o/g tiles for the same (row,col) live in one wave -> fused epilogue.
//
// Data movement (CDNA5 path): the 16 KB B-tile for K-step kb+1 is staged into
// a double-buffered LDS region with global_load_async_to_lds_b128 (ASYNCcnt)
// *before* the 16 v_wmma_f32_16x16x32_bf16 of step kb execute; the wave only
// executes s_wait_asynccnt 0 + barrier after compute, hiding global latency.
// A fragments are software-pipelined in registers the same way.
// ---------------------------------------------------------------------------
__global__ __launch_bounds__(256) void step_gemm_lstm(
    const unsigned short* __restrict__ Xbf,     // (N,T,D) bf16
    const unsigned short* __restrict__ Hin,     // (N,H) bf16  (h_{t-1})
    const unsigned short* __restrict__ Attn,    // (N,H) bf16
    const unsigned short* __restrict__ Wsw,     // swizzled bf16 weights
    const float* __restrict__ bias,             // (4H,)
    float* __restrict__ Cst,                    // (N,H) fp32 cell state
    float* __restrict__ Hprev,                  // (N,H) fp32 h out
    unsigned short* __restrict__ Hout,          // (N,H) bf16 h out (ping-pong)
    float* __restrict__ out,                    // (N,T,H)
    int t)
{
    __shared__ __align__(16) unsigned short Bbuf[2][16 * FRAG_ELEMS]; // 2x16KB

    const int tid  = threadIdx.x;
    const int lane = tid & 31;
    const int wave = tid >> 5;
    const int m0   = blockIdx.x * 128 + wave * 16;   // batch rows of this wave
    const int cb0  = blockIdx.y * 64;                // H-cols of this block
    const int half = lane >> 4;                      // A-layout half select
    const int row  = m0 + (lane & 15);               // A-operand row (this lane)
    const int by   = blockIdx.y;

    const size_t xrow = ((size_t)row * TT + t) * DD;
    const size_t hrow = (size_t)row * HH;

    v8f acc[4][4];
#pragma unroll
    for (int q = 0; q < 4; ++q)
#pragma unroll
        for (int cf = 0; cf < 4; ++cf) { v8f z = {}; acc[q][cf] = z; }

    // cooperative async stage of the 16KB B tile of K-step kb into Bbuf[buf]:
    // per gate q the 4 needed fragments are 4KB contiguous in Wsw; each of the
    // 256 threads moves one 16B chunk per gate (4 async b128 per thread).
    auto stageB = [&](int kb, int buf) {
#pragma unroll
        for (int q = 0; q < 4; ++q) {
            const unsigned short* src =
                Wsw + ((size_t)kb * 256 + (size_t)q * 64 + (size_t)by * 4) *
                          FRAG_ELEMS + (size_t)tid * 8;
            unsigned dst =
                (unsigned)(size_t)(const void*)&Bbuf[buf][q * 2048 + tid * 8];
            async_copy_b128(dst, (unsigned long long)(size_t)src);
        }
    };

    // A fragment: 16x32 bf16, rows m0..m0+15, K block kb (contiguous k order, same as Wsw B)
    auto loadA = [&](int kb, BFrag& a) {
        const unsigned short* ap;
        if (kb < 16)       ap = Xbf  + xrow + (kb << 5);
        else if (kb < 48)  ap = Hin  + hrow + ((kb - 16) << 5);
        else               ap = Attn + hrow + ((kb - 48) << 5);
        ap += half * 16;                             // lanes>=16: K {16..31}
        a.q[0] = *(const u32x4*)(ap);                // K +0..7
        a.q[1] = *(const u32x4*)(ap + 8);            // K +8..15
    };

    // prologue: prime buffer 0 and first A fragment
    stageB(0, 0);
    BFrag aCur;
    loadA(0, aCur);
    wait_asynccnt0();
    __syncthreads();

    for (int kb = 0; kb < NKB; ++kb) {
        const int cur = kb & 1;
        const int nxt = cur ^ 1;
        // issue next tile's async copy + next A load before compute
        if (kb + 1 < NKB) stageB(kb + 1, nxt);
        BFrag aNxt;
        if (kb + 1 < NKB) loadA(kb + 1, aNxt);

        const unsigned short* bl = &Bbuf[cur][lane * 16];
#pragma unroll
        for (int q = 0; q < 4; ++q) {
#pragma unroll
            for (int cf = 0; cf < 4; ++cf) {
                const unsigned short* bp = bl + (q * 4 + cf) * FRAG_ELEMS;
                BFrag b_;
                b_.q[0] = *(const u32x4*)(bp);       // ds_load_b128
                b_.q[1] = *(const u32x4*)(bp + 8);
                acc[q][cf] = __builtin_amdgcn_wmma_f32_16x16x32_bf16(
                    false, aCur.bf, false, b_.bf, (short)0, acc[q][cf],
                    false, false);
            }
        }
        if (kb + 1 < NKB) {
            wait_asynccnt0();       // next tile landed in LDS
            __syncthreads();        // all waves done reading Bbuf[cur]
            aCur = aNxt;
        }
    }

    // ---- fused LSTM epilogue ----
    // C/D layout: VGPR r -> M = r (lanes 0-15) / M = r+8 (lanes 16-31), N = lane&15
#pragma unroll
    for (int cf = 0; cf < 4; ++cf) {
        const int col = cb0 + cf * 16 + (lane & 15);
        const float bi = bias[col];
        const float bf = bias[HH + col];
        const float bo = bias[2 * HH + col];
        const float bg = bias[3 * HH + col];
#pragma unroll
        for (int r = 0; r < 8; ++r) {
            const int orow = m0 + r + half * 8;
            const size_t idx = (size_t)orow * HH + col;
            float iv = acc[0][cf][r] + bi;
            float fv = acc[1][cf][r] + bf;
            float ov = acc[2][cf][r] + bo;
            float gv = acc[3][cf][r] + bg;
            float cp = Cst[idx];
            float cn = sigm(fv) * cp + sigm(iv) * tanhf(gv);
            float hn = sigm(ov) * tanhf(cn);
            Cst[idx]   = cn;
            Hprev[idx] = hn;
            Hout[idx]  = f2bf(hn);
            out[((size_t)orow * TT + t) * HH + col] = hn;
        }
    }
}

// ---------------------------------------------------------------------------
extern "C" void kernel_launch(void* const* d_in, const int* in_sizes, int n_in,
                              void* d_out, int out_size, void* d_ws, size_t ws_size,
                              hipStream_t stream)
{
    (void)in_sizes; (void)n_in; (void)out_size; (void)ws_size;
    const float* x     = (const float*)d_in[0];   // (N,T,D)
    const float* A     = (const float*)d_in[1];   // (N,H,4,4)
    const float* Wx    = (const float*)d_in[2];   // (D,4H)
    const float* Wh    = (const float*)d_in[3];   // (H,4H)
    const float* Wattn = (const float*)d_in[4];   // (H,4H)
    const float* b     = (const float*)d_in[5];   // (4H,)
    float* out = (float*)d_out;                   // (N,T,H)

    char* ws = (char*)d_ws;
    size_t off = 0;
    auto alloc = [&](size_t bytes) -> void* {
        void* p = ws + off;
        off += (bytes + 255) & ~(size_t)255;
        return p;
    };
    unsigned short* Wsw    = (unsigned short*)alloc((size_t)KTOT * NCOL * 2);  // 20 MB
    unsigned short* Xbf    = (unsigned short*)alloc((size_t)NB * TT * DD * 2); // 32 MB
    unsigned short* HbfA   = (unsigned short*)alloc((size_t)NB * HH * 2);
    unsigned short* HbfB   = (unsigned short*)alloc((size_t)NB * HH * 2);
    unsigned short* Attnbf = (unsigned short*)alloc((size_t)NB * HH * 2);
    float*          Hprev  = (float*)alloc((size_t)NB * HH * 4);
    float*          Cst    = (float*)alloc((size_t)NB * HH * 4);

    // one-time conversions (recomputed every call -> deterministic)
    swz_w_kernel<<<(KTOT * NCOL) / 256, 256, 0, stream>>>(Wx, Wh, Wattn, Wsw);
    conv_x_kernel<<<(NB * TT * DD) / 256, 256, 0, stream>>>(x, Xbf, NB * TT * DD);
    init_h0_kernel<<<(NB * HH) / 256, 256, 0, stream>>>(A, Hprev, Cst, HbfA);

    dim3 gg(2, 16);  // 2 x 128 batch rows, 16 x 64 H-cols
    for (int t = 0; t < TT; ++t) {
        const unsigned short* hin = (t & 1) ? HbfB : HbfA;
        unsigned short*       hout = (t & 1) ? HbfA : HbfB;
        attn_kernel<<<NB, 256, 0, stream>>>(A, Hprev, Attnbf);
        step_gemm_lstm<<<gg, 256, 0, stream>>>(Xbf, hin, Attnbf, Wsw, b,
                                               Cst, Hprev, hout, out, t);
    }
}
